// VectorQuantizer_13159779794957
// MI455X (gfx1250) — matrix operand order repaired and checked by reading, not compiled, not run on hardware
//
#include <hip/hip_runtime.h>

// ---------------------------------------------------------------------------
// VQ-VAE vector quantizer for MI455X (gfx1250, wave32, WMMA + TDM).
//   prep kernel:  codebook fp32 -> f16 (packed) + ||e||^2 norms, once, in d_ws
//   main kernel:  TDM tensor_load_to_lds stages f16 codebook into padded LDS
//                 (pad_enable inserts the 8-half row pad -> 72-half stride),
//                 scores = z_tile(f16) @ codebook^T(f16) via
//                 v_wmma_f32_16x16x32_f16, argmin fused in-register,
//                 exact fp32 gather + loss afterwards.
// ---------------------------------------------------------------------------

typedef _Float16 half_t;
typedef __attribute__((ext_vector_type(16))) _Float16 v16h;
typedef __attribute__((ext_vector_type(8)))  _Float16 v8h;
typedef __attribute__((ext_vector_type(8)))  float    v8f;
typedef unsigned int u32x4 __attribute__((ext_vector_type(4)));
typedef int          i32x8 __attribute__((ext_vector_type(8)));
typedef int          i32x4 __attribute__((ext_vector_type(4)));

#define K_CODES     512
#define CDIM        64
#define ROWS_PER_WG 128
#define CB_STRIDE   72      // halves; 144B rows -> 16B aligned, bank-conflict-free
#define Z_STRIDE    72
#define HW          4096    // 64*64 spatial per image
#define BSTRIDE     262144  // 64*4096 elements per image
#define NTOTAL      8388608 // 32*64*64*64

// d_ws layout (bytes):
//   [0,      65536)  f16 codebook, packed [512][64]
//   [65536,  67584)  fp32 norms  [512]
//   [67584,  71680)  fp32 per-WG loss partials [1024]

// ---- prep: fp32 codebook -> f16 + norms of the f16-rounded rows ----
__global__ __launch_bounds__(64)
void vq_prep_13159779794957(const float* __restrict__ cb,
                            half_t* __restrict__ cb16,
                            float* __restrict__ norms)
{
    __shared__ float red[64];
    const int code = blockIdx.x;
    const int c    = threadIdx.x;
    float  v = cb[code * CDIM + c];
    half_t h = (half_t)v;
    cb16[code * CDIM + c] = h;
    float e = (float)h;
    red[c] = e * e;
    __syncthreads();
    for (int s = 32; s > 0; s >>= 1) {
        if (c < s) red[c] += red[c + s];
        __syncthreads();
    }
    if (c == 0) norms[code] = red[0];
}

__global__ __launch_bounds__(256)
void vq_main_13159779794957(const float* __restrict__ z,
                            const float* __restrict__ cb,
                            const half_t* __restrict__ cb16,
                            const float* __restrict__ norms,
                            float* __restrict__ out,
                            float* __restrict__ partial)
{
    __shared__ __align__(16) half_t lds_cb[K_CODES * CB_STRIDE];    // 73728 B
    __shared__ __align__(16) half_t lds_z [ROWS_PER_WG * Z_STRIDE]; // 18432 B
    __shared__ float lds_norm[K_CODES];
    __shared__ int   s_idx[ROWS_PER_WG];
    __shared__ float s_red[256];

    const int t  = threadIdx.x;
    const int wg = blockIdx.x;
    const int b  = wg >> 5;                 // 32 WGs per image
    const int s0 = (wg & 31) * ROWS_PER_WG; // spatial offset inside image
    const size_t zbase = (size_t)b * BSTRIDE + (size_t)s0;

#if __has_builtin(__builtin_amdgcn_tensor_load_to_lds)
    // ---- TDM: DMA the f16 codebook into LDS with hardware row padding ----
    // D# group0: count=1 | lds_addr | global_addr(57b) | type=2
    // D# group1: data_size=2B, pad_enable, pad_interval=32 dwords (64 halves),
    //            pad_amount=4 dwords (8 halves) -> LDS row stride 72 halves.
    //            tensor 64x512, tile 64x512, dim0 stride 64.
    if (t < 32) {   // wave 0 issues; TDM ignores EXEC, one op per wave
        unsigned           lds_off = (unsigned)(size_t)(&lds_cb[0]);
        unsigned long long ga      = (unsigned long long)(size_t)cb16;
        u32x4 g0 = { 1u,
                     lds_off,
                     (unsigned)(ga & 0xffffffffu),
                     (unsigned)((ga >> 32) & 0x1ffffffu) | 0x80000000u };
        i32x8 g1 = { (int)((1u << 16) | (1u << 20) | (4u << 22) | (3u << 25)),
                     (int)((unsigned)CDIM << 16),        // tensor_dim0 = 64
                     (int)((unsigned)K_CODES << 16),     // tensor_dim1 = 512
                     (int)((unsigned)CDIM << 16),        // tile_dim0   = 64
                     (int)K_CODES,                       // tile_dim1   = 512
                     (int)CDIM,                          // dim0 stride = 64
                     0, 0 };
        i32x4 g2 = { 0, 0, 0, 0 };
        i32x4 g3 = { 0, 0, 0, 0 };
#if defined(__clang_major__) && (__clang_major__ >= 23)
        i32x8 g4 = { 0, 0, 0, 0, 0, 0, 0, 0 };
        __builtin_amdgcn_tensor_load_to_lds(g0, g1, g2, g3, g4, 0);
#else
        __builtin_amdgcn_tensor_load_to_lds(g0, g1, g2, g3, 0);
#endif
    }
#else
    // ---- fallback: vectorized copy of precomputed f16 codebook into LDS ----
    {
        const v8h* src = (const v8h*)cb16;   // 4096 16B chunks
        #pragma unroll
        for (int i = 0; i < 16; ++i) {
            int idx  = i * 256 + t;
            int code = idx >> 3;             // 8 chunks per code row
            int j    = idx & 7;
            *(v8h*)&lds_cb[code * CB_STRIDE + j * 8] = src[idx];
        }
    }
#endif

    // ---- stage z tile (fp32 [C][S] -> f16 [row][c]) while TDM runs ----
    {
        const float4* z4 = (const float4*)(z + zbase); // 16B aligned (s0 % 128 == 0)
        #pragma unroll
        for (int i = 0; i < 8; ++i) {
            int idx4 = i * 256 + t;                  // 2048 float4 total
            int c = idx4 >> 5;                       // 32 float4 per channel
            int j = idx4 & 31;
            float4 v = z4[c * (HW / 4) + j];
            int row = j * 4;
            lds_z[(row + 0) * Z_STRIDE + c] = (half_t)v.x;
            lds_z[(row + 1) * Z_STRIDE + c] = (half_t)v.y;
            lds_z[(row + 2) * Z_STRIDE + c] = (half_t)v.z;
            lds_z[(row + 3) * Z_STRIDE + c] = (half_t)v.w;
        }
    }
    // ---- stage precomputed norms ----
    lds_norm[t]       = norms[t];
    lds_norm[t + 256] = norms[t + 256];

#if __has_builtin(__builtin_amdgcn_tensor_load_to_lds)
    if (t < 32) __builtin_amdgcn_s_wait_tensorcnt(0);
#endif
    __syncthreads();

    // ---- per-wave 16x512 score block via WMMA, fused argmin ----
    const int wave = t >> 5;
    const int lane = t & 31;
    const int hi   = lane >> 4;   // lane half selects K sub-group
    const int ln   = lane & 15;   // A: row index / B: column index

    // A fragments (16-bit A 16x32 layout, ISA 7.12.2), persistent for all tiles
    const half_t* arow = &lds_z[(wave * 16 + ln) * Z_STRIDE + hi * 8];
    v16h a0, a1;
    {
        v8h lo0 = *(const v8h*)(arow +  0);   // K = hi*8 + 0..7
        v8h hi0 = *(const v8h*)(arow + 16);   // K = hi*8 + 16..23
        v8h lo1 = *(const v8h*)(arow + 32);   // K = 32 + hi*8 + 0..7
        v8h hi1 = *(const v8h*)(arow + 48);   // K = 32 + hi*8 + 16..23
        #pragma unroll
        for (int j = 0; j < 8; ++j) {
            a0[j] = lo0[j]; a0[j + 8] = hi0[j];
            a1[j] = lo1[j]; a1[j + 8] = hi1[j];
        }
    }

    float bestd[8];
    int   bestn[8];
    #pragma unroll
    for (int i = 0; i < 8; ++i) { bestd[i] = 3.4028235e38f; bestn[i] = 0; }

    for (int tile = 0; tile < 32; ++tile) {
        // B fragment (32x16): lane half hi holds K = hi*8 + 0..7 and 16 + hi*8 + 0..7
        const half_t* brow = &lds_cb[(tile * 16 + ln) * CB_STRIDE + hi * 8];
        v16h b0, b1;
        {
            v8h l0 = *(const v8h*)(brow +  0);
            v8h h0 = *(const v8h*)(brow + 16);
            v8h l1 = *(const v8h*)(brow + 32);
            v8h h1 = *(const v8h*)(brow + 48);
            #pragma unroll
            for (int j = 0; j < 8; ++j) {
                b0[j] = l0[j]; b0[j + 8] = h0[j];
                b1[j] = l1[j]; b1[j + 8] = h1[j];
            }
        }
        v8f acc = {};
        acc = __builtin_amdgcn_wmma_f32_16x16x32_f16(false, a0, false, b0,
                                                     (short)0, acc, false, false);
        acc = __builtin_amdgcn_wmma_f32_16x16x32_f16(false, a1, false, b1,
                                                     (short)0, acc, false, false);
        const int   n  = tile * 16 + ln;
        const float nn = lds_norm[n];
        #pragma unroll
        for (int i = 0; i < 8; ++i) {       // dist' = ||e||^2 - 2 x.e
            float d = fmaf(-2.f, acc[i], nn);
            bool better = (d < bestd[i]);
            bestd[i] = better ? d : bestd[i];
            bestn[i] = better ? n : bestn[i];
        }
    }

    // cross-lane (value,index) min-reduce within each 16-lane half
    #pragma unroll
    for (int mask = 1; mask < 16; mask <<= 1) {
        #pragma unroll
        for (int i = 0; i < 8; ++i) {
            float od = __shfl_xor(bestd[i], mask, 32);
            int   on = __shfl_xor(bestn[i], mask, 32);
            bool take = (od < bestd[i]) || (od == bestd[i] && on < bestn[i]);
            bestd[i] = take ? od : bestd[i];
            bestn[i] = take ? on : bestn[i];
        }
    }
    if (ln == 0) {   // D-layout: VGPR i <-> row (i + 8*hi)
        #pragma unroll
        for (int i = 0; i < 8; ++i)
            s_idx[wave * 16 + hi * 8 + i] = bestn[i];
    }
    __syncthreads();

    // ---- exact fp32 gather -> output, plus exact loss accumulation ----
    float acc_l = 0.f;
    #pragma unroll
    for (int i = 0; i < 32; ++i) {
        int idx = i * 256 + t;               // 8192 outputs for this WG
        int c   = idx >> 7;
        int sl  = idx & 127;
        int k   = s_idx[sl];
        float q  = cb[k * CDIM + c];         // L2-hot gather (128 KB codebook)
        size_t ga = zbase + (size_t)c * HW + (size_t)sl;
        float zv = z[ga];
        out[ga] = q;                         // straight-through forward == q
        float d = q - zv;
        acc_l = fmaf(d, d, acc_l);
    }
    s_red[t] = acc_l;
    __syncthreads();
    for (int s = 128; s > 0; s >>= 1) {
        if (t < s) s_red[t] += s_red[t + s];
        __syncthreads();
    }
    if (t == 0) partial[wg] = s_red[0];
}

__global__ __launch_bounds__(256)
void vq_loss_reduce_13159779794957(const float* __restrict__ partial,
                                   float* __restrict__ out)
{
    __shared__ float s_red[256];
    const int t = threadIdx.x;
    float s = partial[t] + partial[t + 256] + partial[t + 512] + partial[t + 768];
    s_red[t] = s;
    __syncthreads();
    for (int k = 128; k > 0; k >>= 1) {
        if (t < k) s_red[t] += s_red[t + k];
        __syncthreads();
    }
    // loss = q_latent + 0.25*e_latent = 1.25 * mean((q - z)^2)
    if (t == 0) out[NTOTAL] = 1.25f * s_red[0] / (float)NTOTAL;
}

extern "C" void kernel_launch(void* const* d_in, const int* in_sizes, int n_in,
                              void* d_out, int out_size, void* d_ws, size_t ws_size,
                              hipStream_t stream)
{
    const float* z  = (const float*)d_in[0];   // [32,64,64,64] fp32
    const float* cb = (const float*)d_in[1];   // [512,64] fp32
    float* out      = (float*)d_out;           // 8388608 quantized + 1 loss

    char*   ws      = (char*)d_ws;             // needs >= 71680 bytes
    half_t* cb16    = (half_t*)ws;
    float*  norms   = (float*)(ws + 65536);
    float*  partial = (float*)(ws + 67584);

    vq_prep_13159779794957<<<dim3(K_CODES), dim3(64), 0, stream>>>(cb, cb16, norms);
    vq_main_13159779794957<<<dim3(1024), dim3(256), 0, stream>>>(z, cb, cb16, norms,
                                                                 out, partial);
    vq_loss_reduce_13159779794957<<<dim3(1), dim3(256), 0, stream>>>(partial, out);
}
